// CrossChannelBlock1_83047487636079
// MI455X (gfx1250) — matrix operand order repaired and checked by reading, not compiled, not run on hardware
//
#include <hip/hip_runtime.h>
#include <hip/hip_bf16.h>

typedef unsigned int u32;
typedef __attribute__((ext_vector_type(16))) _Float16 v16h;
typedef __attribute__((ext_vector_type(8)))  float    v8f;
typedef __attribute__((ext_vector_type(8)))  unsigned int u32x8;

#define BDIM 4
#define HDIM 128
#define WDIM 128
#define NTOK (HDIM * WDIM)      // 16384
#define CDIM 256
#define NH   8
#define HD   32
#define MROWS (BDIM * NTOK)     // 65536

__device__ __forceinline__ float gelu_exact(float x) {
    return 0.5f * x * (1.0f + erff(x * 0.70710678118654752f));
}

// ---------------------------------------------------------------------------
// Weight pack: f32 [K][N] row-major -> u32 [K/2][N], each u32 = {f16(W[2k2][n]) lo, f16(W[2k2+1][n]) hi}
// Matches the (k,k+1) 16-bit pair packing the WMMA B-fragment expects.
// ---------------------------------------------------------------------------
__global__ void pack_w16_kernel(const float* __restrict__ w, u32* __restrict__ out, int K2, int N) {
    int i = blockIdx.x * 256 + threadIdx.x;
    if (i >= K2 * N) return;
    int k2 = i / N, n = i - k2 * N;
    _Float16 lo = (_Float16)w[(size_t)(2 * k2) * N + n];
    _Float16 hi = (_Float16)w[(size_t)(2 * k2 + 1) * N + n];
    u32 v = ((u32)__builtin_bit_cast(unsigned short, hi) << 16) |
            (u32)__builtin_bit_cast(unsigned short, lo);
    out[i] = v;
}

// ---------------------------------------------------------------------------
// Depthwise 3x3 conv PE (+residual) with optional fused LayerNorm -> f16 out.
// block = 256 threads = one (b, token)'s channel row. grid = B*NTOK.
// ---------------------------------------------------------------------------
__global__ __launch_bounds__(256) void dwconv_ln_kernel(
    const float* __restrict__ src, const float* __restrict__ w9,
    const float* __restrict__ cb,
    const float* __restrict__ lnw, const float* __restrict__ lnb,
    float* __restrict__ dst, _Float16* __restrict__ ln16)
{
    __shared__ float r1[256];
    __shared__ float r2[256];
    int b = blockIdx.x >> 14;
    int n = blockIdx.x & (NTOK - 1);
    int h = n >> 7, x = n & (WDIM - 1);
    int c = threadIdx.x;
    const float* base = src + (size_t)b * NTOK * CDIM;
    float acc = cb[c];
    for (int dy = -1; dy <= 1; ++dy) {
        int hy = h + dy;
        if (hy < 0 || hy >= HDIM) continue;
        for (int dx = -1; dx <= 1; ++dx) {
            int xx = x + dx;
            if (xx < 0 || xx >= WDIM) continue;
            acc += base[(size_t)(hy * WDIM + xx) * CDIM + c] * w9[c * 9 + (dy + 1) * 3 + (dx + 1)];
        }
    }
    float val = base[(size_t)n * CDIM + c] + acc;
    dst[((size_t)b * NTOK + n) * CDIM + c] = val;

    if (ln16 != nullptr) {
        r1[c] = val; r2[c] = val * val;
        __syncthreads();
        for (int s = 128; s > 0; s >>= 1) {
            if (c < s) { r1[c] += r1[c + s]; r2[c] += r2[c + s]; }
            __syncthreads();
        }
        float mean = r1[0] * (1.0f / 256.0f);
        float var  = r2[0] * (1.0f / 256.0f) - mean * mean;
        float inv  = rsqrtf(var + 1e-5f);
        float o = (val - mean) * inv * lnw[c] + lnb[c];
        ln16[((size_t)b * NTOK + n) * CDIM + c] = (_Float16)o;
    }
}

// ---------------------------------------------------------------------------
// WMMA GEMM: out[M,N] = act(A[M,K] @ W[K,N] + bias) (+ residual)
// A: f16 row-major. Bp: packed u32 [K/2][N]. Block tile 128x128, BK=32.
// 8 waves (4 along M x 2 along N); per-wave 32x64 = 2x4 v_wmma_f32_16x16x32_f16.
// ---------------------------------------------------------------------------
__global__ __launch_bounds__(256) void gemm_wmma_kernel(
    const _Float16* __restrict__ A, const u32* __restrict__ Bp,
    const float* __restrict__ bias, const float* residual,
    float* outF32, _Float16* outF16,
    int M, int N, int K, int doGelu)
{
    __shared__ u32 Alds[128][17];    // 128 rows x 16 u32 (32 f16), pad 1
    __shared__ u32 Blds[16][132];    // 16 pair-rows x 128 cols, pad 4

    const int tid  = threadIdx.x;
    const int lane = tid & 31;
    const int wid  = tid >> 5;
    const int wm   = wid & 3;        // 4 waves along M
    const int wn   = wid >> 2;       // 2 waves along N (64 cols each)
    const int m0   = blockIdx.y * 128;
    const int n0   = blockIdx.x * 128;
    const int K2   = K >> 1;
    const u32* Au  = (const u32*)A;

    v8f acc[2][4];
    const v8f z8 = {0.f, 0.f, 0.f, 0.f, 0.f, 0.f, 0.f, 0.f};
#pragma unroll
    for (int mi = 0; mi < 2; ++mi)
#pragma unroll
        for (int ni = 0; ni < 4; ++ni) acc[mi][ni] = z8;

    const int ar = tid >> 2;            // 0..63
    const int ac = (tid & 3) << 2;      // 0,4,8,12
    const int br = tid >> 4;            // 0..15
    const int bc = (tid & 15) << 2;     // 0..60

    for (int kb = 0; kb < K; kb += 32) {
        const int kc = (kb >> 1);
        // stage A tile (128 x 16 u32)
        {
            const uint4 v0 = *(const uint4*)(Au + (size_t)(m0 + ar) * K2 + kc + ac);
            Alds[ar][ac + 0] = v0.x; Alds[ar][ac + 1] = v0.y;
            Alds[ar][ac + 2] = v0.z; Alds[ar][ac + 3] = v0.w;
            const uint4 v1 = *(const uint4*)(Au + (size_t)(m0 + ar + 64) * K2 + kc + ac);
            Alds[ar + 64][ac + 0] = v1.x; Alds[ar + 64][ac + 1] = v1.y;
            Alds[ar + 64][ac + 2] = v1.z; Alds[ar + 64][ac + 3] = v1.w;
        }
        // stage B tile (16 x 128 u32)
        {
            const uint4 v0 = *(const uint4*)(Bp + (size_t)(kc + br) * N + n0 + bc);
            Blds[br][bc + 0] = v0.x; Blds[br][bc + 1] = v0.y;
            Blds[br][bc + 2] = v0.z; Blds[br][bc + 3] = v0.w;
            const uint4 v1 = *(const uint4*)(Bp + (size_t)(kc + br) * N + n0 + 64 + bc);
            Blds[br][bc + 64] = v1.x; Blds[br][bc + 65] = v1.y;
            Blds[br][bc + 66] = v1.z; Blds[br][bc + 67] = v1.w;
        }
        if (kb + 32 < K) {
            __builtin_prefetch(Au + (size_t)(m0 + ar) * K2 + kc + 16 + ac, 0, 1);
            __builtin_prefetch(Bp + (size_t)(kc + 16 + br) * N + n0 + bc, 0, 1);
            __builtin_prefetch(Bp + (size_t)(kc + 16 + br) * N + n0 + 64 + bc, 0, 1);
        }
        __syncthreads();

        // load fragments per CDNA5 16-bit A(16x32)/B(32x16) layouts
        v16h af[2], bf[4];
        const int abase = (lane < 16) ? 0 : 4;  // lanes>=16 hold K=8..15 / 24..31
        const int mrow0 = wm * 32 + (lane & 15);
#pragma unroll
        for (int mi = 0; mi < 2; ++mi) {
            u32x8 t;
            const int mr = mrow0 + mi * 16;
#pragma unroll
            for (int j = 0; j < 4; ++j) {
                t[j]     = Alds[mr][abase + j];
                t[4 + j] = Alds[mr][abase + 8 + j];
            }
            af[mi] = __builtin_bit_cast(v16h, t);
        }
        const int rbase = (lane < 16) ? 0 : 4;  // lanes>=16 hold K=8..15 / 24..31
        const int ncol0 = wn * 64 + (lane & 15);
#pragma unroll
        for (int ni = 0; ni < 4; ++ni) {
            u32x8 t;
            const int nc = ncol0 + ni * 16;
#pragma unroll
            for (int j = 0; j < 8; ++j) t[j] = Blds[rbase + j + ((j >> 2) << 2)][nc];
            bf[ni] = __builtin_bit_cast(v16h, t);
        }
        __syncthreads();

#pragma unroll
        for (int mi = 0; mi < 2; ++mi)
#pragma unroll
            for (int ni = 0; ni < 4; ++ni)
                acc[mi][ni] = __builtin_amdgcn_wmma_f32_16x16x32_f16(
                    false, af[mi], false, bf[ni], (short)0, acc[mi][ni], false, false);
    }

    // epilogue: C layout -> lane: col = n&15 (+16-group), VGPR j: row j (+8 for lanes>=16)
    const int rowTop = m0 + wm * 32 + ((lane >> 4) << 3);
    const int col0   = n0 + wn * 64 + (lane & 15);
#pragma unroll
    for (int mi = 0; mi < 2; ++mi) {
#pragma unroll
        for (int ni = 0; ni < 4; ++ni) {
            const int col = col0 + ni * 16;
            float bv = (bias != nullptr) ? bias[col] : 0.0f;
#pragma unroll
            for (int j = 0; j < 8; ++j) {
                int row = rowTop + mi * 16 + j;
                float v = acc[mi][ni][j] + bv;
                if (doGelu) v = gelu_exact(v);
                size_t idx = (size_t)row * N + col;
                if (outF16 != nullptr) outF16[idx] = (_Float16)v;
                if (outF32 != nullptr) outF32[idx] = (residual != nullptr) ? residual[idx] + v : v;
            }
        }
    }
}

// ---------------------------------------------------------------------------
// XCA logits: S[b,h,d,e] += scale * sum_n k[n,d]*v[n,e]. grid = B*NH*8 splits.
// lane = e; k broadcast over d by shuffles. LDS reduce, then global f32 atomics.
// ---------------------------------------------------------------------------
__global__ __launch_bounds__(256) void xca_logits_kernel(
    const _Float16* __restrict__ kv, float* __restrict__ S)
{
    __shared__ float sacc[32][33];
    int blk = blockIdx.x;
    int b = blk >> 6;
    int h = (blk >> 3) & 7;
    int split = blk & 7;
    int tid = threadIdx.x, lane = tid & 31, wid = tid >> 5;
    int t0 = split * 2048 + wid * 256;

    float acc[32];
#pragma unroll
    for (int d = 0; d < 32; ++d) acc[d] = 0.0f;

    for (int t = 0; t < 256; ++t) {
        const _Float16* row = kv + ((size_t)(b * NTOK + t0 + t)) * (2 * CDIM) + h * HD;
        float kval = (float)row[lane];          // k[n, d=lane]
        float vval = (float)row[CDIM + lane];   // v[n, e=lane]
#pragma unroll
        for (int d = 0; d < 32; ++d)
            acc[d] += __shfl(kval, d, 32) * vval;
    }

    for (int i = tid; i < 32 * 33; i += 256) (&sacc[0][0])[i] = 0.0f;
    __syncthreads();
#pragma unroll
    for (int d = 0; d < 32; ++d) atomicAdd(&sacc[d][lane], acc[d]);
    __syncthreads();

    const float scale = 0.17677669529663687f;   // 32^-0.5 applied to k
    for (int i = tid; i < 1024; i += 256) {
        int d = i >> 5, e = i & 31;
        atomicAdd(&S[((size_t)(b * NH + h) * 32 + d) * 32 + e], sacc[d][e] * scale);
    }
}

// softmax over e (32 lanes), write transposed attnT[bh][e][d]
__global__ __launch_bounds__(256) void softmax_attn_kernel(
    const float* __restrict__ S, float* __restrict__ attnT)
{
    int tid = threadIdx.x, lane = tid & 31, wid = tid >> 5;
    int r = blockIdx.x * 8 + wid;           // 0..B*NH*32-1
    int bh = r >> 5, d = r & 31;
    float v = S[((size_t)bh * 32 + d) * 32 + lane];
    float m = v;
#pragma unroll
    for (int o = 16; o > 0; o >>= 1) m = fmaxf(m, __shfl_xor(m, o, 32));
    float e = __expf(v - m);
    float s = e;
#pragma unroll
    for (int o = 16; o > 0; o >>= 1) s += __shfl_xor(s, o, 32);
    attnT[((size_t)bh * 32 + lane) * 32 + d] = e / s;
}

// out[n, h*32+d] = sum_e attn[d,e]*q[n,e];  attn matrices for batch b in LDS.
__global__ __launch_bounds__(256) void apply_attn_kernel(
    const _Float16* __restrict__ q, const float* __restrict__ attnT,
    _Float16* __restrict__ outh)
{
    __shared__ float at[NH * 32 * 32];      // 32 KB
    int b = blockIdx.x >> 8;
    int chunk = blockIdx.x & 255;
    int tid = threadIdx.x, lane = tid & 31, wid = tid >> 5;
    for (int i = tid; i < NH * 1024; i += 256) at[i] = attnT[(size_t)b * NH * 1024 + i];
    __syncthreads();
    int tbase = chunk * 64 + wid * 8;
    for (int tt = 0; tt < 8; ++tt) {
        size_t rowoff = ((size_t)(b * NTOK + tbase + tt)) * CDIM;
        for (int h = 0; h < NH; ++h) {
            float qv = (float)q[rowoff + h * HD + lane];
            float o = 0.0f;
#pragma unroll
            for (int e = 0; e < 32; ++e)
                o += at[h * 1024 + e * 32 + lane] * __shfl(qv, e, 32);
            outh[rowoff + h * HD + lane] = (_Float16)o;
        }
    }
}

__global__ void zero_f32_kernel(float* p, int n) {
    int i = blockIdx.x * 256 + threadIdx.x;
    if (i < n) p[i] = 0.0f;
}

// mean-pool over tokens: atomic accumulate per (b, c). grid = B*16 chunks.
__global__ __launch_bounds__(256) void pool_kernel(
    const float* __restrict__ src, float* __restrict__ pooled, int outOff)
{
    int b = blockIdx.x >> 4;
    int chunk = blockIdx.x & 15;
    int c = threadIdx.x;
    const float* base = src + ((size_t)b * NTOK + chunk * 1024) * CDIM + c;
    float s = 0.0f;
    for (int t = 0; t < 1024; ++t) s += base[(size_t)t * CDIM];
    atomicAdd(&pooled[b * (2 * CDIM) + outOff + c], s);
}

// tiny gating MLP: g = sigmoid(gelu(pooled/NTOK @ w1 + b1) @ w2 + b2), per batch.
__global__ __launch_bounds__(256) void gate_mlp_kernel(
    const float* __restrict__ pooled,
    const float* __restrict__ w1, const float* __restrict__ b1,
    const float* __restrict__ w2, const float* __restrict__ b2,
    float* __restrict__ g)
{
    __shared__ float pl[512];
    __shared__ float g1[256];
    int tid = threadIdx.x;
    for (int b = 0; b < BDIM; ++b) {
        pl[tid]       = pooled[b * 512 + tid]       * (1.0f / (float)NTOK);
        pl[tid + 256] = pooled[b * 512 + 256 + tid] * (1.0f / (float)NTOK);
        __syncthreads();
        float a1 = b1[tid];
        for (int k = 0; k < 512; ++k) a1 += pl[k] * w1[k * 256 + tid];
        g1[tid] = gelu_exact(a1);
        __syncthreads();
        for (int jj = 0; jj < 2; ++jj) {
            int j = tid + jj * 256;
            float a2 = b2[j];
            for (int c2 = 0; c2 < 256; ++c2) a2 += g1[c2] * w2[c2 * 512 + j];
            g[b * 512 + j] = 1.0f / (1.0f + __expf(-a2));
        }
        __syncthreads();
    }
}

// f = s_x*y + s_y*x  (s_x gates y!), keep f32 residual + LN2 -> f16.
__global__ __launch_bounds__(256) void fuse_ln_kernel(
    const float* __restrict__ x3, const float* __restrict__ y3,
    const float* __restrict__ g,
    const float* __restrict__ lnw, const float* __restrict__ lnb,
    float* __restrict__ ft, _Float16* __restrict__ ln16)
{
    __shared__ float r1[256];
    __shared__ float r2[256];
    int b = blockIdx.x >> 14;
    int n = blockIdx.x & (NTOK - 1);
    int c = threadIdx.x;
    size_t idx = ((size_t)b * NTOK + n) * CDIM + c;
    float val = g[b * 512 + c] * y3[idx] + g[b * 512 + 256 + c] * x3[idx];
    ft[idx] = val;
    r1[c] = val; r2[c] = val * val;
    __syncthreads();
    for (int s = 128; s > 0; s >>= 1) {
        if (c < s) { r1[c] += r1[c + s]; r2[c] += r2[c + s]; }
        __syncthreads();
    }
    float mean = r1[0] * (1.0f / 256.0f);
    float var  = r2[0] * (1.0f / 256.0f) - mean * mean;
    float inv  = rsqrtf(var + 1e-5f);
    ln16[idx] = (_Float16)((val - mean) * inv * lnw[c] + lnb[c]);
}

// (B, N, C) f32 -> (B, C, N) via 32x32 LDS tiles
__global__ __launch_bounds__(256) void transpose_out_kernel(
    const float* __restrict__ ft, float* __restrict__ out)
{
    __shared__ float tile[32][33];
    int blk = blockIdx.x;
    int b  = blk >> 12;             // / (512*8)
    int nb = (blk >> 3) & 511;
    int cb = blk & 7;
    int n0 = nb * 32, c0 = cb * 32;
    int tx = threadIdx.x & 31, ty = threadIdx.x >> 5;
    for (int k = 0; k < 4; ++k) {
        int rr = ty + k * 8;
        tile[rr][tx] = ft[((size_t)b * NTOK + n0 + rr) * CDIM + c0 + tx];
    }
    __syncthreads();
    for (int k = 0; k < 4; ++k) {
        int rr = ty + k * 8;
        out[((size_t)b * CDIM + c0 + rr) * NTOK + n0 + tx] = tile[tx][rr];
    }
}

// ---------------------------------------------------------------------------
extern "C" void kernel_launch(void* const* d_in, const int* in_sizes, int n_in,
                              void* d_out, int out_size, void* d_ws, size_t ws_size,
                              hipStream_t stream) {
    (void)in_sizes; (void)n_in; (void)out_size; (void)ws_size;
    // dict order (setup_inputs):
    const float* x_in    = (const float*)d_in[0];
    const float* y_in    = (const float*)d_in[1];
    const float* cpex0_w = (const float*)d_in[2];  const float* cpex0_b = (const float*)d_in[3];
    const float* cpey0_w = (const float*)d_in[4];  const float* cpey0_b = (const float*)d_in[5];
    const float* cpex1_w = (const float*)d_in[6];  const float* cpex1_b = (const float*)d_in[7];
    const float* cpey1_w = (const float*)d_in[8];  const float* cpey1_b = (const float*)d_in[9];
    const float* ln1x_w  = (const float*)d_in[10]; const float* ln1x_b  = (const float*)d_in[11];
    const float* ln1y_w  = (const float*)d_in[12]; const float* ln1y_b  = (const float*)d_in[13];
    const float* ln2_w   = (const float*)d_in[14]; const float* ln2_b   = (const float*)d_in[15];
    const float* ax_kv_w = (const float*)d_in[16]; const float* ax_q_w  = (const float*)d_in[17];
    const float* ax_p_w  = (const float*)d_in[18]; const float* ax_p_b  = (const float*)d_in[19];
    const float* ay_kv_w = (const float*)d_in[20]; const float* ay_q_w  = (const float*)d_in[21];
    const float* ay_p_w  = (const float*)d_in[22]; const float* ay_p_b  = (const float*)d_in[23];
    const float* ci_w1   = (const float*)d_in[24]; const float* ci_b1   = (const float*)d_in[25];
    const float* ci_w2   = (const float*)d_in[26]; const float* ci_b2   = (const float*)d_in[27];
    const float* fc1_w   = (const float*)d_in[28]; const float* fc1_b   = (const float*)d_in[29];
    const float* fc2_w   = (const float*)d_in[30]; const float* fc2_b   = (const float*)d_in[31];
    float* out = (float*)d_out;

    // workspace layout
    char* wsb = (char*)d_ws;
    size_t off = 0;
    auto take = [&](size_t bytes) -> char* {
        char* p = wsb + off;
        off += (bytes + 255) & ~(size_t)255;
        return p;
    };
    const size_t F32_MNC = (size_t)MROWS * CDIM * 4;   // 64 MB
    const size_t F16_MNC = (size_t)MROWS * CDIM * 2;   // 32 MB

    u32* wkv_x = (u32*)take(128 * 512 * 4);
    u32* wq_x  = (u32*)take(128 * 256 * 4);
    u32* wp_x  = (u32*)take(128 * 256 * 4);
    u32* wkv_y = (u32*)take(128 * 512 * 4);
    u32* wq_y  = (u32*)take(128 * 256 * 4);
    u32* wp_y  = (u32*)take(128 * 256 * 4);
    u32* wfc1  = (u32*)take(128 * 1024 * 4);
    u32* wfc2  = (u32*)take(512 * 256 * 4);
    float* xbuf = (float*)take(F32_MNC);
    float* ybuf = (float*)take(F32_MNC);
    float* x3   = (float*)take(F32_MNC);
    float* y3   = (float*)take(F32_MNC);
    float* ft   = (float*)take(F32_MNC);
    _Float16* curx16 = (_Float16*)take(F16_MNC);
    _Float16* cury16 = (_Float16*)take(F16_MNC);
    char* big16 = take((size_t)MROWS * 1024 * 2);      // 128 MB, also fc1 activations
    _Float16* kv16  = (_Float16*)big16;
    _Float16* q16   = (_Float16*)(big16 + (size_t)MROWS * 512 * 2);
    _Float16* att16 = (_Float16*)(big16 + (size_t)MROWS * 768 * 2);
    _Float16* fc1a16 = (_Float16*)big16;
    float* S      = (float*)take(BDIM * NH * 32 * 32 * 4);
    float* attnT  = (float*)take(BDIM * NH * 32 * 32 * 4);
    float* pooled = (float*)take(BDIM * 512 * 4);
    float* gbuf   = (float*)take(BDIM * 512 * 4);
    _Float16* ln2_16 = curx16;   // reuse (curx' dead by then)

    const int STOT = BDIM * NH * 32 * 32;   // 32768

    // 1) pack weights to f16 pair-packed layout
    pack_w16_kernel<<<(128 * 512 + 255) / 256, 256, 0, stream>>>(ax_kv_w, wkv_x, 128, 512);
    pack_w16_kernel<<<(128 * 256 + 255) / 256, 256, 0, stream>>>(ax_q_w,  wq_x,  128, 256);
    pack_w16_kernel<<<(128 * 256 + 255) / 256, 256, 0, stream>>>(ax_p_w,  wp_x,  128, 256);
    pack_w16_kernel<<<(128 * 512 + 255) / 256, 256, 0, stream>>>(ay_kv_w, wkv_y, 128, 512);
    pack_w16_kernel<<<(128 * 256 + 255) / 256, 256, 0, stream>>>(ay_q_w,  wq_y,  128, 256);
    pack_w16_kernel<<<(128 * 256 + 255) / 256, 256, 0, stream>>>(ay_p_w,  wp_y,  128, 256);
    pack_w16_kernel<<<(128 * 1024 + 255) / 256, 256, 0, stream>>>(fc1_w, wfc1, 128, 1024);
    pack_w16_kernel<<<(512 * 256 + 255) / 256, 256, 0, stream>>>(fc2_w, wfc2, 512, 256);

    // 2) dwconv PE #0 + LN1 (fused)
    dwconv_ln_kernel<<<BDIM * NTOK, 256, 0, stream>>>(x_in, cpex0_w, cpex0_b, ln1x_w, ln1x_b, xbuf, curx16);
    dwconv_ln_kernel<<<BDIM * NTOK, 256, 0, stream>>>(y_in, cpey0_w, cpey0_b, ln1y_w, ln1y_b, ybuf, cury16);

    // 3) x-stream XCA:  kv from cury, q from curx
    gemm_wmma_kernel<<<dim3(512 / 128, MROWS / 128), 256, 0, stream>>>(
        cury16, wkv_x, nullptr, nullptr, nullptr, kv16, MROWS, 512, 256, 0);
    gemm_wmma_kernel<<<dim3(256 / 128, MROWS / 128), 256, 0, stream>>>(
        curx16, wq_x, nullptr, nullptr, nullptr, q16, MROWS, 256, 256, 0);
    zero_f32_kernel<<<(STOT + 255) / 256, 256, 0, stream>>>(S, STOT);
    xca_logits_kernel<<<BDIM * NH * 8, 256, 0, stream>>>(kv16, S);
    softmax_attn_kernel<<<(BDIM * NH * 32) / 8, 256, 0, stream>>>(S, attnT);
    apply_attn_kernel<<<BDIM * 256, 256, 0, stream>>>(q16, attnT, att16);
    // proj: f16 out = curx' (pre-residual, feeds y-XCA kv); f32 out = xbuf + curx'
    gemm_wmma_kernel<<<dim3(256 / 128, MROWS / 128), 256, 0, stream>>>(
        att16, wp_x, ax_p_b, xbuf, xbuf, curx16, MROWS, 256, 256, 0);

    // 4) y-stream XCA: kv from curx' (x-XCA output), q from cury
    gemm_wmma_kernel<<<dim3(512 / 128, MROWS / 128), 256, 0, stream>>>(
        curx16, wkv_y, nullptr, nullptr, nullptr, kv16, MROWS, 512, 256, 0);
    gemm_wmma_kernel<<<dim3(256 / 128, MROWS / 128), 256, 0, stream>>>(
        cury16, wq_y, nullptr, nullptr, nullptr, q16, MROWS, 256, 256, 0);
    zero_f32_kernel<<<(STOT + 255) / 256, 256, 0, stream>>>(S, STOT);
    xca_logits_kernel<<<BDIM * NH * 8, 256, 0, stream>>>(kv16, S);
    softmax_attn_kernel<<<(BDIM * NH * 32) / 8, 256, 0, stream>>>(S, attnT);
    apply_attn_kernel<<<BDIM * 256, 256, 0, stream>>>(q16, attnT, att16);
    gemm_wmma_kernel<<<dim3(256 / 128, MROWS / 128), 256, 0, stream>>>(
        att16, wp_y, ay_p_b, ybuf, ybuf, nullptr, MROWS, 256, 256, 0);

    // 5) dwconv PE #1 (no LN)
    dwconv_ln_kernel<<<BDIM * NTOK, 256, 0, stream>>>(xbuf, cpex1_w, cpex1_b, nullptr, nullptr, x3, nullptr);
    dwconv_ln_kernel<<<BDIM * NTOK, 256, 0, stream>>>(ybuf, cpey1_w, cpey1_b, nullptr, nullptr, y3, nullptr);

    // 6) channel-interaction gating
    zero_f32_kernel<<<(BDIM * 512 + 255) / 256, 256, 0, stream>>>(pooled, BDIM * 512);
    pool_kernel<<<BDIM * 16, 256, 0, stream>>>(x3, pooled, 0);
    pool_kernel<<<BDIM * 16, 256, 0, stream>>>(y3, pooled, 256);
    gate_mlp_kernel<<<1, 256, 0, stream>>>(pooled, ci_w1, ci_b1, ci_w2, ci_b2, gbuf);

    // 7) cross-gated fuse + LN2
    fuse_ln_kernel<<<BDIM * NTOK, 256, 0, stream>>>(x3, y3, gbuf, ln2_w, ln2_b, ft, ln2_16);

    // 8) MLP: fc1 (gelu) then fc2 (+residual ft)
    gemm_wmma_kernel<<<dim3(1024 / 128, MROWS / 128), 256, 0, stream>>>(
        ln2_16, wfc1, fc1_b, nullptr, nullptr, fc1a16, MROWS, 1024, 256, 1);
    gemm_wmma_kernel<<<dim3(256 / 128, MROWS / 128), 256, 0, stream>>>(
        fc1a16, wfc2, fc2_b, ft, ft, nullptr, MROWS, 256, 1024, 0);

    // 9) (B,N,C) -> (B,C,H,W)
    transpose_out_kernel<<<BDIM * 512 * 8, 256, 0, stream>>>(ft, out);
}
